// DecodingCUDAGraphRunner_7533372638000
// MI455X (gfx1250) — matrix operand order repaired and checked by reading, not compiled, not run on hardware
//
#include <hip/hip_runtime.h>
#include <hip/hip_bf16.h>

typedef __attribute__((ext_vector_type(16))) _Float16 v16h;
typedef __attribute__((ext_vector_type(8)))  float    v8f;
typedef __attribute__((ext_vector_type(4)))  float    v4f;

#define BB 4
#define SS 2048
#define HH 16
#define DHH 32
#define DD 512
#define FFF 2048
#define NSPLIT 8
#define SCHUNK 256   // SS / NSPLIT

// ---------------------------------------------------------------------------
// LayerNorm over 4 rows of 512, 256 threads (64 threads per row).
// src/dst are LDS [4][512]; sred >= 256 floats; stat >= 8 floats.
// ---------------------------------------------------------------------------
__device__ __forceinline__ void block_ln512(const float* src, float* dst,
                                            const float* __restrict__ w,
                                            const float* __restrict__ b,
                                            float* sred, float* stat) {
  const int t = threadIdx.x;
  const int row = t >> 6, lt = t & 63;
  float s = 0.f;
  for (int i = lt; i < 512; i += 64) s += src[row * 512 + i];
  sred[t] = s;
  __syncthreads();
  for (int off = 32; off > 0; off >>= 1) {
    if (lt < off) sred[t] += sred[t + off];
    __syncthreads();
  }
  if (lt == 0) stat[row] = sred[row * 64] * (1.f / 512.f);
  __syncthreads();
  const float mean = stat[row];
  float v = 0.f;
  for (int i = lt; i < 512; i += 64) {
    float d = src[row * 512 + i] - mean;
    v += d * d;
  }
  __syncthreads();
  sred[t] = v;
  __syncthreads();
  for (int off = 32; off > 0; off >>= 1) {
    if (lt < off) sred[t] += sred[t + off];
    __syncthreads();
  }
  if (lt == 0) stat[4 + row] = rsqrtf(sred[row * 64] * (1.f / 512.f) + 1e-5f);
  __syncthreads();
  const float rstd = stat[4 + row];
  for (int i = lt; i < 512; i += 64)
    dst[row * 512 + i] = (src[row * 512 + i] - mean) * rstd * w[i] + b[i];
  __syncthreads();
}

// ---------------------------------------------------------------------------
// WMMA GEMV tile: out[0..3][col 0..15] = sIn[4][KTOT] x W[col][KTOT]^T.
// 256 threads = 8 waves, K split across waves, LDS reduce.
// A fragment: 16-bit A 16x32 layout (rows 0..3 real, rest zero).
// B fragment: 16-bit B 32x16 layout (lane&15 = column, 16 consecutive K).
// Result valid on wave 0, lanes 0..15: out4[r] = C[M=r][N=lane].
// ---------------------------------------------------------------------------
template <int KTOT>
__device__ __forceinline__ void gemv_tile(const float* sIn,
                                          const float* __restrict__ W,
                                          int colBase, float* sredW,
                                          float* out4) {
  const int t = threadIdx.x, wave = t >> 5, lane = t & 31;
  const int kslice = KTOT / 8;
  const int kBeg = wave * kslice;
  const int r = lane & 15, halfI = lane >> 4;

  __syncthreads();  // sIn produced by all threads before any wave reads it

  v8f c = {0.f, 0.f, 0.f, 0.f, 0.f, 0.f, 0.f, 0.f};
  for (int k0 = kBeg; k0 < kBeg + kslice; k0 += 32) {
    // ---- A fragment (ISA 7.12.2, 16-bit A 16x32) ----
    v16h a;
    if (r < BB) {
      const float* rowp = sIn + r * KTOT;
#pragma unroll
      for (int e = 0; e < 16; ++e) {
        const int j = e >> 1, bit = e & 1;
        const int kk = k0 + (halfI << 4) + (j << 1) + bit;
        a[e] = (_Float16)rowp[kk];
      }
    } else {
#pragma unroll
      for (int e = 0; e < 16; ++e) a[e] = (_Float16)0.f;
    }
    // ---- B fragment (16-bit B 32x16): 16 consecutive K per lane-half ----
    const long roff = (long)(colBase + r) * KTOT + k0 + (halfI << 4);
    const v4f* p4 = (const v4f*)(W + roff);
    __builtin_prefetch(W + roff + 32, 0, 1);
    v16h b;
#pragma unroll
    for (int q4 = 0; q4 < 4; ++q4) {
      v4f f = p4[q4];
      b[q4 * 4 + 0] = (_Float16)f[0];
      b[q4 * 4 + 1] = (_Float16)f[1];
      b[q4 * 4 + 2] = (_Float16)f[2];
      b[q4 * 4 + 3] = (_Float16)f[3];
    }
    c = __builtin_amdgcn_wmma_f32_16x16x32_f16(false, a, false, b, (short)0, c,
                                               false, false);
  }

#pragma unroll
  for (int e = 0; e < 8; ++e) sredW[(wave * 32 + lane) * 8 + e] = c[e];
  __syncthreads();
  if (wave == 0) {
#pragma unroll
    for (int rr = 0; rr < BB; ++rr) {
      float s = 0.f;
#pragma unroll
      for (int w2 = 0; w2 < 8; ++w2) s += sredW[(w2 * 32 + lane) * 8 + rr];
      out4[rr] = s;
    }
  }
}

// ---------------------------------------------------------------------------
// K1: h = (applyLN ? LN(in) : in);  qkv = h @ ipw^T + ipb
// q -> q_out, k -> knew, v -> vnew; block 0 saves h.
// grid 96 (16 cols each), block 256.
// ---------------------------------------------------------------------------
__global__ void __launch_bounds__(256) qkv_kernel(
    const float* __restrict__ in, int applyLN, const float* __restrict__ lnw,
    const float* __restrict__ lnb, const float* __restrict__ ipw,
    const float* __restrict__ ipb, float* __restrict__ q_out,
    float* __restrict__ knew, float* __restrict__ vnew,
    float* __restrict__ h_save) {
  __shared__ float sIn[2048], sH[2048], sRed[256], sStat[8], sW[2048];
  const int t = threadIdx.x;
  for (int i = t; i < 2048; i += 256) sIn[i] = in[i];
  __syncthreads();
  if (applyLN) {
    block_ln512(sIn, sH, lnw, lnb, sRed, sStat);
  } else {
    for (int i = t; i < 2048; i += 256) sH[i] = sIn[i];
    __syncthreads();
  }
  if (blockIdx.x == 0)
    for (int i = t; i < 2048; i += 256) h_save[i] = sH[i];

  float out4[4];
  const int colBase = blockIdx.x * 16;
  gemv_tile<512>(sH, ipw, colBase, sW, out4);

  const int wave = t >> 5, lane = t & 31;
  if (wave == 0 && lane < 16) {
    const int j = colBase + lane;
    const float bias = ipb[j];
#pragma unroll
    for (int r = 0; r < BB; ++r) {
      float v = out4[r] + bias;
      if (j < 512)        q_out[r * 512 + j] = v;
      else if (j < 1024)  knew [r * 512 + (j - 512)] = v;
      else                vnew [r * 512 + (j - 1024)] = v;
    }
  }
}

// ---------------------------------------------------------------------------
// K2: flash-decode partials. grid = B*H*NSPLIT = 512 blocks, 256 threads.
// Position kv_len comes from knew/vnew (cache is never mutated).
// partials[(b*H+h)*NSPLIT + c] = { m, l, out[32] } (stride 34 floats).
// ---------------------------------------------------------------------------
__global__ void __launch_bounds__(256) attn_kernel(
    const float* __restrict__ q, const float* __restrict__ knew,
    const float* __restrict__ vnew, const float* __restrict__ kcache,
    const float* __restrict__ vcache, const int* __restrict__ kvlen_ptr,
    float* __restrict__ partials) {
  __shared__ float sq[32], sp[256], sred[256], sacc[256];
  const int t = threadIdx.x;
  const int chunk = blockIdx.x & 7;
  const int h = (blockIdx.x >> 3) & 15;
  const int b = blockIdx.x >> 7;
  const int kvlen = *kvlen_ptr;
  const int nvalid = kvlen + 1;
  const int c0 = chunk * SCHUNK;

  if (t < 32) sq[t] = q[b * 512 + h * 32 + t] * 0.17677669529663687f;
  __syncthreads();

  const int s = c0 + t;
  const bool valid = (s < nvalid);
  float score = -1e30f;
  if (valid) {
    const float* kp = (s < kvlen)
                          ? (kcache + (((long)b * SS + s) * HH + h) * DHH)
                          : (knew + b * 512 + h * 32);
    float acc = 0.f;
#pragma unroll
    for (int d4 = 0; d4 < 8; ++d4) {
      v4f kf = ((const v4f*)kp)[d4];
      acc += kf[0] * sq[d4 * 4 + 0] + kf[1] * sq[d4 * 4 + 1] +
             kf[2] * sq[d4 * 4 + 2] + kf[3] * sq[d4 * 4 + 3];
    }
    score = acc;
  }
  sred[t] = score;
  __syncthreads();
  for (int off = 128; off > 0; off >>= 1) {
    if (t < off) sred[t] = fmaxf(sred[t], sred[t + off]);
    __syncthreads();
  }
  const float m = sred[0];
  __syncthreads();
  const float p = valid ? __expf(score - m) : 0.f;
  sp[t] = p;
  sred[t] = p;
  __syncthreads();
  for (int off = 128; off > 0; off >>= 1) {
    if (t < off) sred[t] += sred[t + off];
    __syncthreads();
  }
  const float l = sred[0];

  // weighted V: thread = (d = t&31, sgroup = t>>5), 32 seq each
  const int d = t & 31, sg = t >> 5;
  float acc = 0.f;
  const int sbase = c0 + sg * 32;
  for (int i = 0; i < 32; ++i) {
    const int s2 = sbase + i;
    if (s2 >= nvalid) break;
    const float pv = sp[sg * 32 + i];
    const float* vp = (s2 < kvlen)
                          ? (vcache + (((long)b * SS + s2) * HH + h) * DHH)
                          : (vnew + b * 512 + h * 32);
    acc += pv * vp[d];
  }
  sacc[t] = acc;
  __syncthreads();
  if (t < 32) {
    float o = 0.f;
#pragma unroll
    for (int g = 0; g < 8; ++g) o += sacc[g * 32 + t];
    float* pb = partials + ((long)(b * HH + h) * NSPLIT + chunk) * 34;
    if (t == 0) { pb[0] = m; pb[1] = l; }
    pb[2 + t] = o;
  }
}

// ---------------------------------------------------------------------------
// K3: combine attention partials, out-proj GEMV, residual -> r1.
// grid 32, block 256.
// ---------------------------------------------------------------------------
__global__ void __launch_bounds__(256) proj_kernel(
    const float* __restrict__ partials, const float* __restrict__ h_in,
    const float* __restrict__ ow, const float* __restrict__ ob,
    float* __restrict__ r1) {
  __shared__ float sA[2048], sW[2048], sM[64], sInv[64];
  const int t = threadIdx.x;
  if (t < 64) {
    const int b = t >> 4, h = t & 15;
    const float* pb = partials + ((long)(b * HH + h) * NSPLIT) * 34;
    float M = -1e30f;
    for (int c = 0; c < NSPLIT; ++c) M = fmaxf(M, pb[c * 34]);
    float den = 0.f;
    for (int c = 0; c < NSPLIT; ++c) den += __expf(pb[c * 34] - M) * pb[c * 34 + 1];
    sM[t] = M;
    sInv[t] = 1.f / den;
  }
  __syncthreads();
  for (int idx = t; idx < 2048; idx += 256) {
    const int b = idx >> 9, hd = idx & 511, h = hd >> 5, d = hd & 31;
    const float* pb = partials + ((long)(b * HH + h) * NSPLIT) * 34;
    const float M = sM[b * 16 + h], inv = sInv[b * 16 + h];
    float acc = 0.f;
    for (int c = 0; c < NSPLIT; ++c)
      acc += __expf(pb[c * 34] - M) * pb[c * 34 + 2 + d];
    sA[idx] = acc * inv;
  }

  float out4[4];
  const int colBase = blockIdx.x * 16;
  gemv_tile<512>(sA, ow, colBase, sW, out4);

  const int wave = t >> 5, lane = t & 31;
  if (wave == 0 && lane < 16) {
    const int j = colBase + lane;
    const float bias = ob[j];
#pragma unroll
    for (int r = 0; r < BB; ++r)
      r1[r * 512 + j] = out4[r] + bias + h_in[r * 512 + j];
  }
}

// ---------------------------------------------------------------------------
// K4: h1 = LN1(r1); a = relu(h1 @ W1^T + b1). grid 128, block 256.
// ---------------------------------------------------------------------------
__global__ void __launch_bounds__(256) mlp1_kernel(
    const float* __restrict__ r1, const float* __restrict__ ln1w,
    const float* __restrict__ ln1b, const float* __restrict__ w1,
    const float* __restrict__ b1, float* __restrict__ a_out,
    float* __restrict__ h1_save) {
  __shared__ float sIn[2048], sH[2048], sRed[256], sStat[8], sW[2048];
  const int t = threadIdx.x;
  for (int i = t; i < 2048; i += 256) sIn[i] = r1[i];
  __syncthreads();
  block_ln512(sIn, sH, ln1w, ln1b, sRed, sStat);
  if (blockIdx.x == 0)
    for (int i = t; i < 2048; i += 256) h1_save[i] = sH[i];

  float out4[4];
  const int colBase = blockIdx.x * 16;
  gemv_tile<512>(sH, w1, colBase, sW, out4);

  const int wave = t >> 5, lane = t & 31;
  if (wave == 0 && lane < 16) {
    const int j = colBase + lane;
    const float bias = b1[j];
#pragma unroll
    for (int r = 0; r < BB; ++r)
      a_out[r * FFF + j] = fmaxf(out4[r] + bias, 0.f);
  }
}

// ---------------------------------------------------------------------------
// K5: r2 = h1 + a @ W2^T + b2. grid 32, block 256. K = 2048.
// ---------------------------------------------------------------------------
__global__ void __launch_bounds__(256) mlp2_kernel(
    const float* __restrict__ a, const float* __restrict__ h1,
    const float* __restrict__ w2, const float* __restrict__ b2,
    float* __restrict__ r2) {
  __shared__ float sA[4 * FFF], sW[2048];
  const int t = threadIdx.x;
  for (int i = t; i < 4 * FFF; i += 256) sA[i] = a[i];

  float out4[4];
  const int colBase = blockIdx.x * 16;
  gemv_tile<FFF>(sA, w2, colBase, sW, out4);

  const int wave = t >> 5, lane = t & 31;
  if (wave == 0 && lane < 16) {
    const int j = colBase + lane;
    const float bias = b2[j];
#pragma unroll
    for (int r = 0; r < BB; ++r)
      r2[r * 512 + j] = out4[r] + bias + h1[r * 512 + j];
  }
}

// ---------------------------------------------------------------------------
// K6: out = LN2_last(r2). 1 block, 256 threads.
// ---------------------------------------------------------------------------
__global__ void __launch_bounds__(256) final_ln_kernel(
    const float* __restrict__ r2, const float* __restrict__ lnw,
    const float* __restrict__ lnb, float* __restrict__ out) {
  __shared__ float sIn[2048], sOut[2048], sRed[256], sStat[8];
  const int t = threadIdx.x;
  for (int i = t; i < 2048; i += 256) sIn[i] = r2[i];
  __syncthreads();
  block_ln512(sIn, sOut, lnw, lnb, sRed, sStat);
  for (int i = t; i < 2048; i += 256) out[i] = sOut[i];
}

// ---------------------------------------------------------------------------
extern "C" void kernel_launch(void* const* d_in, const int* in_sizes, int n_in,
                              void* d_out, int out_size, void* d_ws,
                              size_t ws_size, hipStream_t stream) {
  (void)in_sizes; (void)n_in; (void)out_size; (void)ws_size;
  const float* x     = (const float*)d_in[0];
  const float* kcach = (const float*)d_in[1];
  const float* vcach = (const float*)d_in[2];
  const float* ipw   = (const float*)d_in[3];
  const float* ipb   = (const float*)d_in[4];
  const float* ow    = (const float*)d_in[5];
  const float* ob    = (const float*)d_in[6];
  const float* l1w   = (const float*)d_in[7];
  const float* l1b   = (const float*)d_in[8];
  const float* w1    = (const float*)d_in[9];
  const float* b1    = (const float*)d_in[10];
  const float* w2    = (const float*)d_in[11];
  const float* b2    = (const float*)d_in[12];
  const float* l2w   = (const float*)d_in[13];
  const float* l2b   = (const float*)d_in[14];
  const int*   kvlen = (const int*)d_in[15];

  float* ws     = (float*)d_ws;
  float* q      = ws;              // 2048
  float* knew   = ws + 2048;       // 2048
  float* vnew   = ws + 4096;       // 2048
  float* hsave  = ws + 6144;       // 2048
  float* h1save = ws + 8192;       // 2048
  float* r1     = ws + 10240;      // 2048
  float* r2     = ws + 12288;      // 2048
  float* abuf   = ws + 14336;      // 8192
  float* parts  = ws + 22528;      // 4*16*8*34 = 17408

  const size_t cacheL = (size_t)BB * SS * HH * DHH;  // per-layer cache stride

  for (int l = 0; l < 24; ++l) {
    const float* inP = (l == 0) ? x : r2;
    const float* lnw = (l == 0) ? l2w : (l2w + (size_t)(l - 1) * DD);
    const float* lnb = (l == 0) ? l2b : (l2b + (size_t)(l - 1) * DD);
    qkv_kernel<<<96, 256, 0, stream>>>(
        inP, (l > 0) ? 1 : 0, lnw, lnb, ipw + (size_t)l * 3 * DD * DD,
        ipb + (size_t)l * 3 * DD, q, knew, vnew, hsave);
    attn_kernel<<<BB * HH * NSPLIT, 256, 0, stream>>>(
        q, knew, vnew, kcach + (size_t)l * cacheL, vcach + (size_t)l * cacheL,
        kvlen, parts);
    proj_kernel<<<32, 256, 0, stream>>>(
        parts, hsave, ow + (size_t)l * DD * DD, ob + (size_t)l * DD, r1);
    mlp1_kernel<<<128, 256, 0, stream>>>(
        r1, l1w + (size_t)l * DD, l1b + (size_t)l * DD,
        w1 + (size_t)l * FFF * DD, b1 + (size_t)l * FFF, abuf, h1save);
    mlp2_kernel<<<32, 256, 0, stream>>>(
        abuf, h1save, w2 + (size_t)l * DD * FFF, b2 + (size_t)l * DD, r2);
  }
  final_ln_kernel<<<1, 256, 0, stream>>>(r2, l2w + (size_t)23 * DD,
                                         l2b + (size_t)23 * DD, (float*)d_out);
}
